// TransformerLM_26001732010383
// MI455X (gfx1250) — compile-verified, matrix operand order checked by reading, not run on hardware
//
#include <hip/hip_runtime.h>
#include <hip/hip_bf16.h>
#include <math.h>

typedef __attribute__((ext_vector_type(16))) _Float16     v16h;
typedef __attribute__((ext_vector_type(8)))  float        v8f;
typedef __attribute__((ext_vector_type(4)))  unsigned int u32x4;
typedef __attribute__((ext_vector_type(8)))  int          i32x8;
typedef __attribute__((ext_vector_type(4)))  int          i32x4;

#define VSZ   32000
#define EMB   1024
#define NH    16
#define NLAY  4
#define HIDD  4096
#define SEQ   1024
#define BSZ   4
#define HSZ   64
#define NTOK  (BSZ*SEQ)   // 4096 rows in the flattened [B*T, E] activation

#if defined(__has_builtin)
#if __has_builtin(__builtin_amdgcn_tensor_load_to_lds)
#define HAVE_TDM 1
#endif
#endif

static __device__ __forceinline__ v8f v8f_zero() {
  v8f z;
#pragma unroll
  for (int i = 0; i < 8; ++i) z[i] = 0.0f;
  return z;
}

// ---------------------------------------------------------------------------
// Embedding + sinusoidal positional encoding: x[r,:] = tok_emb[idx[r],:]*32 + PE
// ---------------------------------------------------------------------------
__global__ void embed_kernel(const int* __restrict__ idx,
                             const float* __restrict__ tok_emb,
                             float* __restrict__ x) {
  int row = blockIdx.x;                 // 0..NTOK-1
  int t   = row & (SEQ - 1);
  int tok = idx[row];
  const float* src = tok_emb + (size_t)tok * EMB;
  float* dst = x + (size_t)row * EMB;
  const float c0 = 9.210340371976184f / (float)EMB;   // ln(10000)/E
  for (int i = threadIdx.x; i < EMB; i += blockDim.x) {
    float freq = __expf(-(float)(i & ~1) * c0);
    float arg  = (float)t * freq;
    float pe   = (i & 1) ? __cosf(arg) : __sinf(arg);
    dst[i] = src[i] * 32.0f + pe;       // sqrt(1024)=32
  }
}

// ---------------------------------------------------------------------------
// LayerNorm over E=1024 (f32 in, f16 out feeding the WMMA GEMMs)
// ---------------------------------------------------------------------------
__global__ void layernorm_kernel(const float* __restrict__ x,
                                 const float* __restrict__ g,
                                 const float* __restrict__ b,
                                 _Float16* __restrict__ out) {
  int row = blockIdx.x;
  const float* xr = x + (size_t)row * EMB;
  float vals[4], s = 0.f, ss = 0.f;
#pragma unroll
  for (int i = 0; i < 4; ++i) {
    float v = xr[threadIdx.x + i * 256];
    vals[i] = v; s += v; ss += v * v;
  }
  __shared__ float r1[256], r2[256];
  r1[threadIdx.x] = s; r2[threadIdx.x] = ss;
  __syncthreads();
  for (int off = 128; off > 0; off >>= 1) {
    if (threadIdx.x < off) {
      r1[threadIdx.x] += r1[threadIdx.x + off];
      r2[threadIdx.x] += r2[threadIdx.x + off];
    }
    __syncthreads();
  }
  float mean = r1[0] * (1.0f / EMB);
  float var  = r2[0] * (1.0f / EMB) - mean * mean;
  float inv  = rsqrtf(var + 1e-5f);
  _Float16* orow = out + (size_t)row * EMB;
#pragma unroll
  for (int i = 0; i < 4; ++i) {
    int c = threadIdx.x + i * 256;
    orow[c] = (_Float16)((vals[i] - mean) * inv * g[c] + b[c]);
  }
}

// ---------------------------------------------------------------------------
// WMMA GEMM: C[M,N] = act( A[M,K](f16) @ W(f32) + bias + resid )
// BL=0: W row-major [K,N].  BL=1: W is wq/wk/wv layout [H,E,64], col n=h*64+hs.
// OUTF16: write f16 (activations) vs f32 (+residual) output.
// Block tile 128x128x32, 256 thr = 8 waves (4x2), wave tile 32x64.
// A tile staged to LDS by the Tensor Data Mover (wave 0 issues, TENSORcnt).
// ---------------------------------------------------------------------------
template <int BL, int OUTF16>
__launch_bounds__(256)
__global__ void gemm_wmma_kernel(const _Float16* __restrict__ A,
                                 const float* __restrict__ W,
                                 const float* __restrict__ bias,
                                 const float* __restrict__ resid,
                                 void* __restrict__ Cout,
                                 int M, int N, int K, int relu) {
  // 32 halfs/row + 2-DWORD TDM pad -> 36-half stride (18 dwords, conflict-free)
  __shared__ _Float16 As[128][36];      // [m][k]
  __shared__ _Float16 Bs[128][34];      // [n][k] (transposed for paired k loads)
  const int tid  = threadIdx.x;
  const int m0   = blockIdx.y * 128;
  const int n0   = blockIdx.x * 128;
  const int wave = tid >> 5, lane = tid & 31;
  const int wm = wave >> 1, wn = wave & 1;
  const int ml = lane & 15;
  const int kb = (lane < 16) ? 0 : 8;

  v8f acc[2][4];
#pragma unroll
  for (int a = 0; a < 2; ++a)
#pragma unroll
    for (int n = 0; n < 4; ++n) acc[a][n] = v8f_zero();

  const int brow = tid >> 3, bnseg = (tid & 7) * 16;

  for (int k0 = 0; k0 < K; k0 += 32) {
    __syncthreads();
#if defined(HAVE_TDM)
    // --- A tile via Tensor Data Mover: 128x32 f16, row stride K, LDS-padded ---
    if (tid < 32) {                     // wave 0 only; EXEC all-ones
      unsigned long long ga =
          (unsigned long long)(__SIZE_TYPE__)(A + (size_t)m0 * K + k0);
      u32x4 g0 = {1u,                                  // count=1 (valid D#)
                  (unsigned)(__SIZE_TYPE__)&As[0][0],  // lds_addr (byte offset)
                  (unsigned)(ga & 0xffffffffu),
                  (unsigned)((ga >> 32) & 0x1ffffffu) | (2u << 30)};  // type=2
      i32x8 g1 = {(int)((1u << 16) |    // data_size = 2 bytes
                        (1u << 20) |    // pad_enable
                        (3u << 22) |    // pad_interval: 16 DWORDs (one row)
                        (1u << 25)),    // pad_amount: 2 DWORDs
                  (int)(32u << 16),     // tensor_dim0 = 32 (low16 @ bits63:48)
                  (int)(128u << 16),    // tensor_dim1 = 128 (low16 @ bits95:80)
                  (int)(32u << 16),     // tile_dim0 = 32 (bits127:112)
                  (int)128,             // tile_dim1 = 128, tile_dim2 = 0
                  (int)K,               // tensor_dim0_stride = K elements
                  0, 0};
      i32x4 g2 = {0, 0, 0, 0};
      i32x4 g3 = {0, 0, 0, 0};
      i32x8 g4 = {0, 0, 0, 0, 0, 0, 0, 0};
      __builtin_amdgcn_tensor_load_to_lds(g0, g1, g2, g3, g4, 0);
    }
#else
    {
      const int arow2 = tid >> 1, acseg2 = (tid & 1) * 16;
      const _Float16* ag = A + (size_t)(m0 + arow2) * K + k0 + acseg2;
#pragma unroll
      for (int i = 0; i < 16; ++i) As[arow2][acseg2 + i] = ag[i];
    }
#endif
    // --- B tile: f32 weights -> f16 LDS (transposed) ---
#pragma unroll
    for (int i = 0; i < 16; ++i) {
      int n = bnseg + i, gn = n0 + n;
      float vb;
      if (BL == 0) vb = W[(size_t)(k0 + brow) * N + gn];
      else         vb = W[((size_t)(gn >> 6) * K + (k0 + brow)) * 64 + (gn & 63)];
      Bs[n][brow] = (_Float16)vb;
    }
    // prefetch next k-tile of W into L2
    if (k0 + 32 < K) {
      if (BL == 0) __builtin_prefetch(&W[(size_t)(k0 + 32 + brow) * N + n0 + bnseg], 0, 0);
      else __builtin_prefetch(
          &W[((size_t)((n0 + bnseg) >> 6) * K + (k0 + 32 + brow)) * 64 + ((n0 + bnseg) & 63)], 0, 0);
    }
#if defined(HAVE_TDM)
    if (tid < 32) __builtin_amdgcn_s_wait_tensorcnt((short)0);
#endif
    __syncthreads();

    v16h af[2];
#pragma unroll
    for (int mt = 0; mt < 2; ++mt) {
      int r = wm * 32 + mt * 16 + ml;
#pragma unroll
      for (int v = 0; v < 8; ++v) {
        int kk = (v < 4) ? (kb + 2 * v) : (16 + kb + 2 * (v - 4));
        af[mt][2 * v]     = As[r][kk];
        af[mt][2 * v + 1] = As[r][kk + 1];
      }
    }
#pragma unroll
    for (int nt = 0; nt < 4; ++nt) {
      v16h bf;
      int cn = wn * 64 + nt * 16 + ml;
#pragma unroll
      for (int v = 0; v < 8; ++v) {
        int kk = (v < 4) ? (kb + 2 * v) : (16 + kb + 2 * (v - 4));
        bf[2 * v]     = Bs[cn][kk];
        bf[2 * v + 1] = Bs[cn][kk + 1];
      }
#pragma unroll
      for (int mt = 0; mt < 2; ++mt)
        acc[mt][nt] = __builtin_amdgcn_wmma_f32_16x16x32_f16(
            false, af[mt], false, bf, (short)0, acc[mt][nt], false, false);
    }
  }

#pragma unroll
  for (int mt = 0; mt < 2; ++mt)
#pragma unroll
    for (int nt = 0; nt < 4; ++nt) {
      int col   = n0 + wn * 64 + nt * 16 + ml;
      int rbase = m0 + wm * 32 + mt * 16 + 8 * (lane >> 4);
      float bv = bias ? bias[col] : 0.0f;
#pragma unroll
      for (int r = 0; r < 8; ++r) {
        int row = rbase + r;
        float vv = acc[mt][nt][r] + bv;
        if (resid) vv += resid[(size_t)row * N + col];
        if (relu)  vv = fmaxf(vv, 0.0f);
        if (OUTF16) ((_Float16*)Cout)[(size_t)row * N + col] = (_Float16)vv;
        else        ((float*)Cout)[(size_t)row * N + col] = vv;
      }
    }
}

// ---------------------------------------------------------------------------
// Flash attention: one wave per (b, h, 16-row q tile). q/k/v/o are [B*T, H*64]
// f16; online softmax in f32; WMMA for both Q@K^T and P@V.
// ---------------------------------------------------------------------------
__launch_bounds__(32)
__global__ void attention_kernel(const _Float16* __restrict__ q,
                                 const _Float16* __restrict__ kx,
                                 const _Float16* __restrict__ vx,
                                 _Float16* __restrict__ o) {
  __shared__ _Float16 Pl[16][34];
  const int bh = blockIdx.x;
  const int bb = bh >> 4, hh = bh & 15;
  const int m0 = blockIdx.y * 16;
  const int lane = threadIdx.x;
  const int ml = lane & 15;
  const int kb = (lane < 16) ? 0 : 8;
  const size_t rowbase = (size_t)bb * SEQ;
  const int hcol = hh * HSZ;
  const float scale = 0.125f;                 // 64^-0.5
  const int hi = 8 * (lane >> 4);

  // Q A-fragments (16x32 each, covering HS=64) -- f16 paired loads
  v16h qa[2];
  {
    const _Float16* qp = q + (rowbase + m0 + ml) * EMB + hcol;
#pragma unroll
    for (int f = 0; f < 2; ++f)
#pragma unroll
      for (int v = 0; v < 8; ++v) {
        int kk = f * 32 + ((v < 4) ? (kb + 2 * v) : (16 + kb + 2 * (v - 4)));
        qa[f][2 * v]     = qp[kk];
        qa[f][2 * v + 1] = qp[kk + 1];
      }
  }

  float rmax[8], rsum[8];
#pragma unroll
  for (int r = 0; r < 8; ++r) { rmax[r] = -3.0e38f; rsum[r] = 0.0f; }
  v8f oacc[4];
#pragma unroll
  for (int nt = 0; nt < 4; ++nt) oacc[nt] = v8f_zero();

  const int send = m0 + 16;                   // causal bound (exclusive)
  for (int s0 = 0; s0 < send; s0 += 32) {
    // S = Q @ K^T for 32 keys -> two 16x16 C-fragments
    v8f sfr[2];
#pragma unroll
    for (int jt = 0; jt < 2; ++jt) {
      v8f c = v8f_zero();
#pragma unroll
      for (int f = 0; f < 2; ++f) {
        v16h bf;
        const _Float16* kp = kx + (rowbase + s0 + jt * 16 + ml) * EMB + hcol + f * 32;
#pragma unroll
        for (int v = 0; v < 8; ++v) {
          int kk = (v < 4) ? (kb + 2 * v) : (16 + kb + 2 * (v - 4));
          bf[2 * v]     = kp[kk];
          bf[2 * v + 1] = kp[kk + 1];
        }
        c = __builtin_amdgcn_wmma_f32_16x16x32_f16(false, qa[f], false, bf,
                                                   (short)0, c, false, false);
      }
      sfr[jt] = c;
    }

    // online softmax update, write P (f16) to LDS
#pragma unroll
    for (int r = 0; r < 8; ++r) {
      int mrow = m0 + r + hi;
      float v0 = sfr[0][r] * scale, v1 = sfr[1][r] * scale;
      if (s0 + ml      > mrow) v0 = -1e30f;
      if (s0 + 16 + ml > mrow) v1 = -1e30f;
      float tmax = fmaxf(v0, v1);
      for (int off = 1; off < 16; off <<= 1)
        tmax = fmaxf(tmax, __shfl_xor(tmax, off, 32));
      float nmax  = fmaxf(rmax[r], tmax);
      float alpha = __expf(rmax[r] - nmax);
      rmax[r] = nmax;
      float p0 = __expf(v0 - nmax), p1 = __expf(v1 - nmax);
      float ps = p0 + p1;
      for (int off = 1; off < 16; off <<= 1)
        ps += __shfl_xor(ps, off, 32);
      rsum[r] = rsum[r] * alpha + ps;
#pragma unroll
      for (int nt = 0; nt < 4; ++nt) oacc[nt][r] *= alpha;
      Pl[r + hi][ml]      = (_Float16)p0;
      Pl[r + hi][16 + ml] = (_Float16)p1;
    }
    __syncthreads();

    // P as A-fragment, V as B-fragments; O += P @ V
    v16h pa;
#pragma unroll
    for (int v = 0; v < 8; ++v) {
      int kk = (v < 4) ? (kb + 2 * v) : (16 + kb + 2 * (v - 4));
      pa[2 * v]     = Pl[ml][kk];
      pa[2 * v + 1] = Pl[ml][kk + 1];
    }
#pragma unroll
    for (int nt = 0; nt < 4; ++nt) {
      v16h vbf;
#pragma unroll
      for (int v = 0; v < 8; ++v) {
        int kk = (v < 4) ? (kb + 2 * v) : (16 + kb + 2 * (v - 4));
        vbf[2 * v]     = vx[(rowbase + s0 + kk)     * EMB + hcol + nt * 16 + ml];
        vbf[2 * v + 1] = vx[(rowbase + s0 + kk + 1) * EMB + hcol + nt * 16 + ml];
      }
      oacc[nt] = __builtin_amdgcn_wmma_f32_16x16x32_f16(false, pa, false, vbf,
                                                        (short)0, oacc[nt], false, false);
    }
    __syncthreads();
  }

#pragma unroll
  for (int nt = 0; nt < 4; ++nt)
#pragma unroll
    for (int r = 0; r < 8; ++r) {
      int row = m0 + r + hi;
      o[(rowbase + row) * EMB + hcol + nt * 16 + ml] = (_Float16)(oacc[nt][r] / rsum[r]);
    }
}

// ---------------------------------------------------------------------------
// Loss: per-row log-softmax over V=32000, mean NLL -> atomicAdd
// ---------------------------------------------------------------------------
__global__ void zero_loss_kernel(float* loss) {
  if (threadIdx.x == 0 && blockIdx.x == 0) *loss = 0.0f;
}

__global__ void loss_kernel(const float* __restrict__ logits,
                            const int* __restrict__ targets,
                            float* __restrict__ loss) {
  int row = blockIdx.x;
  const float* lr = logits + (size_t)row * VSZ;
  __shared__ float sm[256];
  float m = -3.0e38f;
  for (int c = threadIdx.x; c < VSZ; c += 256) m = fmaxf(m, lr[c]);
  sm[threadIdx.x] = m; __syncthreads();
  for (int off = 128; off > 0; off >>= 1) {
    if (threadIdx.x < off) sm[threadIdx.x] = fmaxf(sm[threadIdx.x], sm[threadIdx.x + off]);
    __syncthreads();
  }
  m = sm[0]; __syncthreads();
  float s = 0.0f;
  for (int c = threadIdx.x; c < VSZ; c += 256) s += __expf(lr[c] - m);
  sm[threadIdx.x] = s; __syncthreads();
  for (int off = 128; off > 0; off >>= 1) {
    if (threadIdx.x < off) sm[threadIdx.x] += sm[threadIdx.x + off];
    __syncthreads();
  }
  if (threadIdx.x == 0) {
    float lse = m + logf(sm[0]);
    float lp  = lr[targets[row]] - lse;
    atomicAdd(loss, -lp * (1.0f / (float)NTOK));
  }
}

// ---------------------------------------------------------------------------
extern "C" void kernel_launch(void* const* d_in, const int* in_sizes, int n_in,
                              void* d_out, int out_size, void* d_ws, size_t ws_size,
                              hipStream_t stream) {
  (void)in_sizes; (void)n_in; (void)out_size; (void)ws_size;
  const int*   idx     = (const int*)  d_in[0];
  const int*   targets = (const int*)  d_in[1];
  const float* tok_emb = (const float*)d_in[2];
  const float* ln1_g   = (const float*)d_in[3];
  const float* ln1_b   = (const float*)d_in[4];
  const float* wq      = (const float*)d_in[5];
  const float* wk      = (const float*)d_in[6];
  const float* wv      = (const float*)d_in[7];
  const float* wo      = (const float*)d_in[8];
  const float* bo      = (const float*)d_in[9];
  const float* ln2_g   = (const float*)d_in[10];
  const float* ln2_b   = (const float*)d_in[11];
  const float* w1      = (const float*)d_in[12];
  const float* b1      = (const float*)d_in[13];
  const float* w2      = (const float*)d_in[14];
  const float* b2      = (const float*)d_in[15];
  const float* lnf_g   = (const float*)d_in[16];
  const float* lnf_b   = (const float*)d_in[17];
  const float* lm_w    = (const float*)d_in[18];
  const float* lm_b    = (const float*)d_in[19];

  float* logits = (float*)d_out;
  float* loss   = logits + (size_t)NTOK * VSZ;

  const size_t S = (size_t)NTOK * EMB;          // 4M elements
  float*    x   = (float*)d_ws;                 // 16 MB f32 residual stream
  _Float16* h16 = (_Float16*)(x + S);           // 8 MB LN output
  _Float16* qb  = h16 + S;                      // 8 MB
  _Float16* kbf = qb + S;                       // 8 MB
  _Float16* vbf = kbf + S;                      // 8 MB
  _Float16* ob  = vbf + S;                      // 8 MB
  _Float16* hid = qb;   // MLP hidden [4096,4096] f16 aliases q..o (exactly 4*S)

  dim3 b256(256), b32(32);
  dim3 g_e(EMB / 128, NTOK / 128);
  dim3 g_h(HIDD / 128, NTOK / 128);
  dim3 g_v(VSZ / 128, NTOK / 128);
  dim3 g_at(BSZ * NH, SEQ / 16);

  embed_kernel<<<NTOK, b256, 0, stream>>>(idx, tok_emb, x);
  for (int l = 0; l < NLAY; ++l) {
    const size_t wqk = (size_t)l * NH * EMB * HSZ;   // = l * 1M
    layernorm_kernel<<<NTOK, b256, 0, stream>>>(x, ln1_g + l * EMB, ln1_b + l * EMB, h16);
    gemm_wmma_kernel<1, 1><<<g_e, b256, 0, stream>>>(h16, wq + wqk, nullptr, nullptr, qb,  NTOK, EMB, EMB, 0);
    gemm_wmma_kernel<1, 1><<<g_e, b256, 0, stream>>>(h16, wk + wqk, nullptr, nullptr, kbf, NTOK, EMB, EMB, 0);
    gemm_wmma_kernel<1, 1><<<g_e, b256, 0, stream>>>(h16, wv + wqk, nullptr, nullptr, vbf, NTOK, EMB, EMB, 0);
    attention_kernel<<<g_at, b32, 0, stream>>>(qb, kbf, vbf, ob);
    gemm_wmma_kernel<0, 0><<<g_e, b256, 0, stream>>>(ob, wo + (size_t)l * EMB * EMB,
                                                     bo + l * EMB, x, x, NTOK, EMB, EMB, 0);
    layernorm_kernel<<<NTOK, b256, 0, stream>>>(x, ln2_g + l * EMB, ln2_b + l * EMB, h16);
    gemm_wmma_kernel<0, 1><<<g_h, b256, 0, stream>>>(h16, w1 + (size_t)l * EMB * HIDD,
                                                     b1 + l * HIDD, nullptr, hid, NTOK, HIDD, EMB, 1);
    gemm_wmma_kernel<0, 0><<<g_e, b256, 0, stream>>>(hid, w2 + (size_t)l * HIDD * EMB,
                                                     b2 + l * EMB, x, x, NTOK, EMB, HIDD, 0);
  }
  layernorm_kernel<<<NTOK, b256, 0, stream>>>(x, lnf_g, lnf_b, h16);
  gemm_wmma_kernel<0, 0><<<g_v, b256, 0, stream>>>(h16, lm_w, lm_b, nullptr, logits, NTOK, VSZ, EMB, 0);
  zero_loss_kernel<<<1, b32, 0, stream>>>(loss);
  loss_kernel<<<NTOK, b256, 0, stream>>>(logits, targets, loss);
}
